// UniPhyBlock_11553462026657
// MI455X (gfx1250) — matrix operand order repaired and checked by reading, not compiled, not run on hardware
//
#include <hip/hip_runtime.h>
#include <hip/hip_bf16.h>
#include <math.h>

// ---------------------------------------------------------------------------
// UniPhyBlock forward for MI455X (gfx1250, wave32, WMMA + TDM).
// GEMM stages: v_wmma_f32_16x16x32_bf16 (bf16 in, f32 accum).
// GEMM A tiles: tensor_load_to_lds (TDM) -> LDS, shared by all 8 waves.
// ---------------------------------------------------------------------------

typedef __attribute__((ext_vector_type(16))) __bf16 v16bf;
typedef __attribute__((ext_vector_type(8)))  __bf16 v8bf;
typedef __attribute__((ext_vector_type(8)))  float  v8f;
typedef __attribute__((ext_vector_type(4)))  unsigned int u32x4;
typedef __attribute__((ext_vector_type(8)))  unsigned int u32x8;

#define B_    2
#define T_    16
#define D_    64
#define H_    64
#define W_    64
#define HW_   4096
#define BT_   32
#define C2_   128
#define NTOK_ 131072   // = BT_*HW_ = B_*HW_*T_

__device__ __forceinline__ float gelu_tanh(float x) {
  float x3 = x * x * x;
  return 0.5f * x * (1.f + tanhf(0.79788456080286535f * (x + 0.044715f * x3)));
}

// ======================= weight / constant builders ========================

// DFT matrices: F = exp(-2*pi*i*r*c/64), G = conj(F)/64. Layout: re[4096], im[4096].
__global__ void k_build_fmat(float* FF, float* FG) {
  int i = blockIdx.x * 256 + threadIdx.x;
  if (i >= 4096) return;
  int r = i >> 6, c = i & 63;
  int ph = (r * c) & 63;                       // exact (period 64)
  float th = 6.283185307179586f * (float)ph / 64.f;
  float cs = cosf(th), sn = sinf(th);
  FF[i] = cs;          FF[4096 + i] = -sn;       // F
  FG[i] = cs / 64.f;   FG[4096 + i] = sn / 64.f; // conj(F)/64
}

// Conv weight as row-major [K=1152][N=128]: Wfull = [[wr,-wi],[wi,wr]] per tap.
__global__ void k_build_wconv(const float* wr, const float* wi, float* Wkn) {
  int i = blockIdx.x * 256 + threadIdx.x;
  if (i >= 1152 * 128) return;
  int n = i & 127;            // out channel
  int k = i >> 7;             // tap*128 + ic
  int tap = k >> 7, ic = k & 127;
  int ky = tap / 3, kx = tap % 3;
  float v;
  if (n < 64) v = (ic < 64) ?  wr[((n * 64 + ic) * 3 + ky) * 3 + kx]
                            : -wi[((n * 64 + (ic - 64)) * 3 + ky) * 3 + kx];
  else        v = (ic < 64) ?  wi[(((n - 64) * 64 + ic) * 3 + ky) * 3 + kx]
                            :  wr[(((n - 64) * 64 + (ic - 64)) * 3 + ky) * 3 + kx];
  Wkn[(size_t)k * 128 + n] = v;
}

// Complex 64x64 matrix (row-major re/im) -> real 128x128 [K][N]:
//   out[o] = sum_i in[i] * Wc[o][i], in/out stored [re(64) | im(64)].
__global__ void k_build_cw(const float* Wr, const float* Wi, float* Wkn) {
  int i = blockIdx.x * 256 + threadIdx.x;
  if (i >= 128 * 128) return;
  int n = i & 127, k = i >> 7;
  float v;
  if (n < 64) v = (k < 64) ?  Wr[n * 64 + k] : -Wi[n * 64 + (k - 64)];
  else        v = (k < 64) ?  Wi[(n - 64) * 64 + k] : Wr[(n - 64) * 64 + (k - 64)];
  Wkn[(size_t)k * 128 + n] = v;
}

// Pack row-major f32 [K][N] into bf16 WMMA B fragments.
// B 16-bit 32x16 layout: lane L -> N = L&15, K = (L>>4)*8 + (e&7) + (e>>3)*16, e in [0,16).
__global__ void k_packB(const float* W, __bf16* Bf, int K, int N) {
  int i = blockIdx.x * 256 + threadIdx.x;
  if (i >= K * N) return;
  int e = i & 15;
  int lane = (i >> 4) & 31;
  int f = i >> 9;
  int nT = N >> 4;
  int nb = f % nT, kb = f / nT;
  int kk = kb * 32 + (lane >> 4) * 8 + (e & 7) + (e >> 3) * 16;
  int nn = nb * 16 + (lane & 15);
  Bf[i] = (__bf16)W[(size_t)kk * N + nn];
}

// Complex Gauss-Jordan inverse of V (64x64), single workgroup, LDS-resident.
__global__ void k_invert(const float* Vr, const float* Vi, float* vinv) {
  extern __shared__ float sm[];
  float* M = sm;              // 64*128 complex (interleaved re,im)
  float* sF = sm + 16384;     // 64 complex factors
  float* sP = sm + 16512;     // pivot reciprocal
  int tid = threadIdx.x;
  for (int i = tid; i < 64 * 128; i += 256) {
    int r = i >> 7, c = i & 127;
    float re, im;
    if (c < 64) { re = Vr[r * 64 + c]; im = Vi[r * 64 + c]; }
    else        { re = (c - 64 == r) ? 1.f : 0.f; im = 0.f; }
    M[i * 2] = re; M[i * 2 + 1] = im;
  }
  __syncthreads();
  for (int p = 0; p < 64; ++p) {
    if (tid == 0) {
      float pr = M[(p * 128 + p) * 2], pi = M[(p * 128 + p) * 2 + 1];
      float d = pr * pr + pi * pi; d = (d < 1e-30f) ? 1e-30f : d;
      sP[0] = pr / d; sP[1] = -pi / d;
    }
    __syncthreads();
    if (tid < 128) {
      int idx = (p * 128 + tid) * 2;
      float ar = M[idx], ai = M[idx + 1];
      float br = sP[0], bi = sP[1];
      M[idx] = ar * br - ai * bi; M[idx + 1] = ar * bi + ai * br;
    }
    __syncthreads();
    if (tid < 64) { sF[tid * 2] = M[(tid * 128 + p) * 2]; sF[tid * 2 + 1] = M[(tid * 128 + p) * 2 + 1]; }
    __syncthreads();
    for (int i = tid; i < 64 * 128; i += 256) {
      int r = i >> 7;
      if (r != p) {
        int c = i & 127;
        float fr = sF[r * 2], fi = sF[r * 2 + 1];
        float pr = M[(p * 128 + c) * 2], pi = M[(p * 128 + c) * 2 + 1];
        M[i * 2]     -= fr * pr - fi * pi;
        M[i * 2 + 1] -= fr * pi + fi * pr;
      }
    }
    __syncthreads();
  }
  for (int i = tid; i < 4096; i += 256) {
    int r = i >> 6, c = i & 63;
    vinv[i]        = M[(r * 128 + 64 + c) * 2];
    vinv[4096 + i] = M[(r * 128 + 64 + c) * 2 + 1];
  }
}

// ============================ LN kernels ===================================

// Spatial complex LN: one thread per pixel (bt,hw); LN across 128 = [re|im].
__global__ void k_norm_spatial(const float* xr, const float* xi,
                               const float* g, const float* b,
                               __bf16* xsbf, float* xsr, float* xsi) {
  int t = blockIdx.x * 256 + threadIdx.x;
  int bt = t >> 12, hw = t & 4095;
  const float* pr = xr + (size_t)bt * D_ * HW_ + hw;
  const float* pi = xi + (size_t)bt * D_ * HW_ + hw;
  float s = 0.f, s2 = 0.f;
  for (int d = 0; d < 64; ++d) {
    float a = pr[d * HW_], c = pi[d * HW_];
    s += a + c; s2 += a * a + c * c;
  }
  float m = s * (1.f / 128.f);
  float v = s2 * (1.f / 128.f) - m * m;
  float rs = rsqrtf(v + 1e-5f);
  __bf16* o = xsbf + (size_t)t * 128;
  for (int d = 0; d < 64; ++d) {
    float nr = (pr[d * HW_] - m) * rs * g[d]      + b[d];
    float ni = (pi[d * HW_] - m) * rs * g[64 + d] + b[64 + d];
    o[d] = (__bf16)nr; o[64 + d] = (__bf16)ni;
    xsr[((size_t)bt * 64 + d) * HW_ + hw] = nr;
    xsi[((size_t)bt * 64 + d) * HW_ + hw] = ni;
  }
}

// Temporal complex LN: token r = (b*HW+hw)*T + t ; reads planar x1, writes bf16 A.
__global__ void k_norm_temporal(const float* x1r, const float* x1i,
                                const float* g, const float* b, __bf16* xtbf) {
  int r = blockIdx.x * 256 + threadIdx.x;
  int t = r & 15, n = r >> 4;
  int hw = n & 4095, bb = n >> 12;
  int bt = bb * 16 + t;
  const float* pr = x1r + (size_t)bt * 64 * HW_ + hw;
  const float* pi = x1i + (size_t)bt * 64 * HW_ + hw;
  float s = 0.f, s2 = 0.f;
  for (int d = 0; d < 64; ++d) {
    float a = pr[d * HW_], c = pi[d * HW_];
    s += a + c; s2 += a * a + c * c;
  }
  float m = s * (1.f / 128.f);
  float rs = rsqrtf(s2 * (1.f / 128.f) - m * m + 1e-5f);
  __bf16* o = xtbf + (size_t)r * 128;
  for (int d = 0; d < 64; ++d) {
    o[d]      = (__bf16)((pr[d * HW_] - m) * rs * g[d]      + b[d]);
    o[64 + d] = (__bf16)((pi[d * HW_] - m) * rs * g[64 + d] + b[64 + d]);
  }
}

// Pool-stage complex LN: token r = bt*HW+hw ; writes f32 NHWC "flat".
__global__ void k_norm_pool(const float* x2r, const float* x2i,
                            const float* g, const float* b, float* flat) {
  int r = blockIdx.x * 256 + threadIdx.x;
  int hw = r & 4095, bt = r >> 12;
  const float* pr = x2r + (size_t)bt * 64 * HW_ + hw;
  const float* pi = x2i + (size_t)bt * 64 * HW_ + hw;
  float s = 0.f, s2 = 0.f;
  for (int d = 0; d < 64; ++d) {
    float a = pr[d * HW_], c = pi[d * HW_];
    s += a + c; s2 += a * a + c * c;
  }
  float m = s * (1.f / 128.f);
  float rs = rsqrtf(s2 * (1.f / 128.f) - m * m + 1e-5f);
  float* o = flat + (size_t)r * 128;
  for (int d = 0; d < 64; ++d) {
    o[d]      = (pr[d * HW_] - m) * rs * g[d]      + b[d];
    o[64 + d] = (pi[d * HW_] - m) * rs * g[64 + d] + b[64 + d];
  }
}

// ============================ WMMA GEMM ====================================

// C[M,N] = act(A[M,K] * B + bias). A row-major bf16; B pre-swizzled fragments.
// Block: 256 threads = 8 waves; each wave owns one 16x16 D tile.
// The 16xK A tile (shared by all waves) is DMA'd to LDS by the TDM
// (tensor_load_to_lds, 2D descriptor) and read back as ds_load fragments.
__global__ void k_gemm(const __bf16* __restrict__ A, const __bf16* __restrict__ Bf,
                       const float* __restrict__ bias,
                       float* __restrict__ Cf, __bf16* __restrict__ Cb,
                       int M, int K, int N, int act) {
  __shared__ __bf16 tileA[16 * 512];           // max K = 512 -> 16 KB
  int lane = threadIdx.x & 31;
  int wave = threadIdx.x >> 5;
  size_t mbase = (size_t)blockIdx.x * 16;
  int nT = N >> 4;
  int nb = blockIdx.y * 8 + wave;

  if (wave == 0) {
    // ---- TDM descriptor (CDNA5 ISA) ----
    unsigned long long gaddr = (unsigned long long)(const void*)(A + mbase * (size_t)K);
    unsigned lds = (unsigned)(size_t)(void*)tileA;  // flat LDS aperture: low 32b = offset
    unsigned uK = (unsigned)K;
    u32x4 g0;
    g0[0] = 1u;                                         // count=1, user mode, no gather
    g0[1] = lds;                                        // lds_addr
    g0[2] = (unsigned)(gaddr & 0xffffffffu);            // global_addr[31:0]
    g0[3] = (unsigned)((gaddr >> 32) & 0x1ffffffu) | (2u << 30); // addr[56:32] | type=2
    u32x8 g1;
    g1[0] = 1u << 16;                                   // wg_mask=0, data_size=1 (2B)
    g1[1] = (uK & 0xffffu) << 16;                       // tensor_dim0[15:0]
    g1[2] = ((uK >> 16) & 0xffffu) | (16u << 16);       // tensor_dim0[31:16] | tensor_dim1=16
    g1[3] = (uK & 0xffffu) << 16;                       // tensor_dim1 hi=0 | tile_dim0=K
    g1[4] = 16u;                                        // tile_dim1=16 rows, tile_dim2=0
    g1[5] = uK;                                         // tensor_dim0_stride = K elements
    g1[6] = 0u;                                         // stride hi | dim1_stride lo
    g1[7] = 0u;
    asm volatile("tensor_load_to_lds %0, %1" :: "s"(g0), "s"(g1) : "memory");
    __builtin_amdgcn_s_wait_tensorcnt(0);
  }
  __syncthreads();

  if (nb < nT) {
    int ksel = (lane >> 4) * 8;     // A 16-bit layout: K-half select per lane pair
    const __bf16* Arow = tileA + (lane & 15) * K;
    v8f acc = {};
    int KB = K >> 5;
    for (int kb = 0; kb < KB; ++kb) {
      int k0 = kb * 32;
      __builtin_prefetch(Bf + (((size_t)(kb + 1) * nT + nb) * 32 + lane) * 16, 0, 1);
      v8bf alo = *(const v8bf*)(Arow + k0 + ksel);        // ds_load_b128
      v8bf ahi = *(const v8bf*)(Arow + k0 + ksel + 16);   // ds_load_b128
      v16bf a;
#pragma unroll
      for (int i = 0; i < 8; ++i) { a[i] = alo[i]; a[8 + i] = ahi[i]; }
      v16bf bfrag = *(const v16bf*)(Bf + (((size_t)kb * nT + nb) * 32 + lane) * 16);
      acc = __builtin_amdgcn_wmma_f32_16x16x32_bf16(false, a, false, bfrag,
                                                    (short)0, acc, false, false);
    }
    int ncol = nb * 16 + (lane & 15);
    int mr0 = (lane >> 4) * 8;
    float bv = bias ? bias[ncol] : 0.f;
#pragma unroll
    for (int r = 0; r < 8; ++r) {
      float v = acc[r] + bv;
      if (act == 1) v = gelu_tanh(v);
      size_t idx = (mbase + mr0 + r) * (size_t)N + ncol;
      if (Cf) Cf[idx] = v;
      if (Cb) Cb[idx] = (__bf16)v;
    }
  }
}

// Implicit-GEMM 3x3 conv, 128 in -> 128 out channels, zero ("SAME") padding.
// Block = 8 waves: M = 16 w-positions, N = 128 out channels. K = 9 taps * 128.
__global__ void k_conv(const __bf16* __restrict__ XS, const __bf16* __restrict__ Bf,
                       const float* __restrict__ br, const float* __restrict__ bi,
                       float* __restrict__ OC) {
  int lane = threadIdx.x & 31;
  int wave = threadIdx.x >> 5;
  int wt = blockIdx.x & 3;
  int h  = (blockIdx.x >> 2) & 63;
  int bt = blockIdx.x >> 8;
  int m = lane & 15;
  int ksel = (lane >> 4) * 8;
  int w = wt * 16 + m;
  v8f acc = {};
  for (int tap = 0; tap < 9; ++tap) {
    int dy = tap / 3 - 1, dx = tap % 3 - 1;
    int hh = h + dy, ww = w + dx;
    bool ok = (hh >= 0) & (hh < 64) & (ww >= 0) & (ww < 64);
    const __bf16* Arow = XS + (((size_t)bt * 64 + (ok ? hh : 0)) * 64 + (ok ? ww : 0)) * 128;
#pragma unroll
    for (int cb = 0; cb < 4; ++cb) {
      v16bf a = {};
      if (ok) {
        v8bf alo = *(const v8bf*)(Arow + cb * 32 + ksel);
        v8bf ahi = *(const v8bf*)(Arow + cb * 32 + ksel + 16);
#pragma unroll
        for (int i = 0; i < 8; ++i) { a[i] = alo[i]; a[8 + i] = ahi[i]; }
      }
      int kb = tap * 4 + cb;
      v16bf bfrag = *(const v16bf*)(Bf + (((size_t)kb * 8 + wave) * 32 + lane) * 16);
      acc = __builtin_amdgcn_wmma_f32_16x16x32_bf16(false, a, false, bfrag,
                                                    (short)0, acc, false, false);
    }
  }
  int ncol = wave * 16 + (lane & 15);
  float bv = (ncol < 64) ? br[ncol] : bi[ncol - 64];
  int mr0 = (lane >> 4) * 8;
#pragma unroll
  for (int r = 0; r < 8; ++r) {
    OC[((size_t)bt * 128 + ncol) * HW_ + h * 64 + wt * 16 + mr0 + r] = acc[r] + bv;
  }
}

// ===================== spectral path (DFT in LDS) ==========================

// Z = G*( (F*(X*F)) .* spec )*G with G = conj(F)/64 -> exact fft2/ifft2 pair.
__global__ void k_fft(float* XSR, float* XSI, const float* FF, const float* FG,
                      const float* specr, const float* speci) {
  extern __shared__ float sm[];
  float* Xr = sm;          float* Xi = sm + 4096;
  float* Tr = sm + 8192;   float* Ti = sm + 12288;
  float* Fr = sm + 16384;  float* Fi = sm + 20480;
  int tid = threadIdx.x;
  int img = blockIdx.x;             // bt*64 + d
  int d = img & 63;
  size_t base = (size_t)img * 4096;
  for (int i = tid; i < 4096; i += 256) {
    Xr[i] = XSR[base + i]; Xi[i] = XSI[base + i];
    Fr[i] = FF[i];         Fi[i] = FF[4096 + i];
  }
  __syncthreads();
  // T = X * F
  for (int o = tid; o < 4096; o += 256) {
    int r = o >> 6, c = o & 63;
    float ar = 0.f, ai = 0.f;
    for (int k = 0; k < 64; ++k) {
      float xr = Xr[(r << 6) + k], xi = Xi[(r << 6) + k];
      float fr = Fr[(k << 6) + c], fi = Fi[(k << 6) + c];
      ar += xr * fr - xi * fi; ai += xr * fi + xi * fr;
    }
    Tr[o] = ar; Ti[o] = ai;
  }
  __syncthreads();
  // Y = F * T  -> X
  for (int o = tid; o < 4096; o += 256) {
    int r = o >> 6, c = o & 63;
    float ar = 0.f, ai = 0.f;
    for (int k = 0; k < 64; ++k) {
      float fr = Fr[(r << 6) + k], fi = Fi[(r << 6) + k];
      float tr = Tr[(k << 6) + c], ti = Ti[(k << 6) + c];
      ar += fr * tr - fi * ti; ai += fr * ti + fi * tr;
    }
    Xr[o] = ar; Xi[o] = ai;
  }
  __syncthreads();
  // Y .*= spec ; F <- G
  for (int o = tid; o < 4096; o += 256) {
    float sr = specr[(size_t)d * 4096 + o], si = speci[(size_t)d * 4096 + o];
    float yr = Xr[o], yi = Xi[o];
    Xr[o] = yr * sr - yi * si; Xi[o] = yr * si + yi * sr;
    Fr[o] = FG[o]; Fi[o] = FG[4096 + o];
  }
  __syncthreads();
  // T = Y' * G
  for (int o = tid; o < 4096; o += 256) {
    int r = o >> 6, c = o & 63;
    float ar = 0.f, ai = 0.f;
    for (int k = 0; k < 64; ++k) {
      float xr = Xr[(r << 6) + k], xi = Xi[(r << 6) + k];
      float fr = Fr[(k << 6) + c], fi = Fi[(k << 6) + c];
      ar += xr * fr - xi * fi; ai += xr * fi + xi * fr;
    }
    Tr[o] = ar; Ti[o] = ai;
  }
  __syncthreads();
  // Z = G * T -> X
  for (int o = tid; o < 4096; o += 256) {
    int r = o >> 6, c = o & 63;
    float ar = 0.f, ai = 0.f;
    for (int k = 0; k < 64; ++k) {
      float fr = Fr[(r << 6) + k], fi = Fi[(r << 6) + k];
      float tr = Tr[(k << 6) + c], ti = Ti[(k << 6) + c];
      ar += fr * tr - fi * ti; ai += fr * ti + fi * tr;
    }
    Xr[o] = ar; Xi[o] = ai;
  }
  __syncthreads();
  for (int i = tid; i < 4096; i += 256) { XSR[base + i] = Xr[i]; XSI[base + i] = Xi[i]; }
}

// ======================= combine / scan / pool =============================

// x1 = g*cliff + (1-g)*spec + x   (planar re/im)
__global__ void k_combine1(const float* xr, const float* xi, const float* OC,
                           const float* SR, const float* SI, const float* gate,
                           float* X1R, float* X1I) {
  size_t i = (size_t)blockIdx.x * 256 + threadIdx.x;   // over BT*D*HW
  int hw = (int)(i & 4095);
  int d  = (int)((i >> 12) & 63);
  int bt = (int)(i >> 18);
  float g = gate[0];
  size_t p = ((size_t)bt * 64 + d) * HW_ + hw;
  float cr = OC[((size_t)bt * 128 + d) * HW_ + hw];
  float ci = OC[((size_t)bt * 128 + 64 + d) * HW_ + hw];
  X1R[p] = g * cr + (1.f - g) * SR[p] + xr[p];
  X1I[p] = g * ci + (1.f - g) * SI[p] + xi[p];
}

// Diagonal linear scan over T=16: h_t = exp(lam*dt_t)*h_{t-1} + x_t.
__global__ void k_scan(const float* XE, const float* dt, const float* omega,
                       const float* gamma_dec, __bf16* HE) {
  int t = blockIdx.x * 256 + threadIdx.x;      // over B*HW*D
  int d = t & 63;
  int n = t >> 6;
  int b = n >> 12;
  float gd = gamma_dec[d];
  float sp = (gd > 20.f) ? gd : log1pf(expf(gd));
  float om = omega[d];
  float hr = 0.f, hi = 0.f;
  for (int tt = 0; tt < 16; ++tt) {
    size_t row = ((size_t)n * 16 + tt) * 128;
    float dtv = dt[b * 16 + tt];
    float dec = expf(-sp * dtv);
    float ar = dec * cosf(om * dtv), ai = dec * sinf(om * dtv);
    float xr = XE[row + d], xi = XE[row + 64 + d];
    float nr = ar * hr - ai * hi + xr;
    float ni = ar * hi + ai * hr + xi;
    hr = nr; hi = ni;
    HE[row + d] = (__bf16)hr; HE[row + 64 + d] = (__bf16)hi;
  }
}

// x2 = x1 + backproj (token order (b,hw,t) -> planar (b,t,d,hw)), in place.
__global__ void k_combine2(const float* BP, float* X1R, float* X1I) {
  size_t i = (size_t)blockIdx.x * 256 + threadIdx.x;
  int hw = (int)(i & 4095);
  int d  = (int)((i >> 12) & 63);
  int bt = (int)(i >> 18);
  int b = bt >> 4, t = bt & 15;
  size_t r = (((size_t)b * 4096 + hw) * 16 + t) * 128;
  size_t p = ((size_t)bt * 64 + d) * HW_ + hw;
  X1R[p] += BP[r + d];
  X1I[p] += BP[r + 64 + d];
}

// mix = avgpool3x3(flat)/9 - flat ; 4 channels per thread.
__global__ void k_mix(const float* flat, __bf16* mix) {
  int t = blockIdx.x * 256 + threadIdx.x;      // over NTOK*32
  int cq = (t & 31) * 4;
  int pix = t >> 5;
  int hw = pix & 4095, bt = pix >> 12;
  int h = hw >> 6, w = hw & 63;
  float a0 = 0.f, a1 = 0.f, a2 = 0.f, a3 = 0.f;
  for (int dy = -1; dy <= 1; ++dy)
    for (int dx = -1; dx <= 1; ++dx) {
      int hh = h + dy, ww = w + dx;
      if (hh < 0 || hh > 63 || ww < 0 || ww > 63) continue;
      const float* p = flat + (((size_t)bt * 4096 + (hh << 6) + ww) * 128 + cq);
      a0 += p[0]; a1 += p[1]; a2 += p[2]; a3 += p[3];
    }
  const float* c = flat + ((size_t)pix * 128 + cq);
  __bf16* o = mix + ((size_t)pix * 128 + cq);
  o[0] = (__bf16)(a0 * (1.f / 9.f) - c[0]);
  o[1] = (__bf16)(a1 * (1.f / 9.f) - c[1]);
  o[2] = (__bf16)(a2 * (1.f / 9.f) - c[2]);
  o[3] = (__bf16)(a3 * (1.f / 9.f) - c[3]);
}

// out = (po complex) + x2, interleaved complex64 into d_out.
__global__ void k_final(const float* PO, const float* X2R, const float* X2I,
                        float* out) {
  size_t i = (size_t)blockIdx.x * 256 + threadIdx.x;
  int hw = (int)(i & 4095);
  int d  = (int)((i >> 12) & 63);
  int bt = (int)(i >> 18);
  size_t r = (size_t)bt * 4096 + hw;
  size_t p = ((size_t)bt * 64 + d) * HW_ + hw;
  out[2 * p]     = PO[r * 128 + d]      + X2R[p];
  out[2 * p + 1] = PO[r * 128 + 64 + d] + X2I[p];
}

// ============================== launcher ===================================

extern "C" void kernel_launch(void* const* d_in, const int* in_sizes, int n_in,
                              void* d_out, int out_size, void* d_ws, size_t ws_size,
                              hipStream_t stream) {
  const float* xr    = (const float*)d_in[0];
  const float* xi    = (const float*)d_in[1];
  const float* dt    = (const float*)d_in[2];
  const float* ns_g  = (const float*)d_in[3];
  const float* ns_b  = (const float*)d_in[4];
  const float* wr    = (const float*)d_in[5];
  const float* wi    = (const float*)d_in[6];
  const float* br    = (const float*)d_in[7];
  const float* bi    = (const float*)d_in[8];
  const float* specr = (const float*)d_in[9];
  const float* speci = (const float*)d_in[10];
  const float* gate  = (const float*)d_in[11];
  const float* nt_g  = (const float*)d_in[12];
  const float* nt_b  = (const float*)d_in[13];
  const float* omega = (const float*)d_in[14];
  const float* gdec  = (const float*)d_in[15];
  const float* V_r   = (const float*)d_in[16];
  const float* V_i   = (const float*)d_in[17];
  const float* np_g  = (const float*)d_in[18];
  const float* np_b  = (const float*)d_in[19];
  const float* w1    = (const float*)d_in[20];
  const float* b1    = (const float*)d_in[21];
  const float* w2    = (const float*)d_in[22];
  const float* b2    = (const float*)d_in[23];
  float* out = (float*)d_out;

  char* ws = (char*)d_ws;
  size_t off = 0;
  auto take = [&](size_t bytes) { size_t o = off; off += (bytes + 255) & ~(size_t)255; return o; };

  __bf16* XSBF  = (__bf16*)(ws + take((size_t)NTOK_ * 128 * 2));   // normed spatial, NHWC bf16
  float*  XSR   = (float*) (ws + take((size_t)BT_ * 64 * HW_ * 4)); // normed re (then spectral out)
  float*  XSI   = (float*) (ws + take((size_t)BT_ * 64 * HW_ * 4));
  float*  FF    = (float*) (ws + take(8192 * 4));
  float*  FG    = (float*) (ws + take(8192 * 4));
  float*  WCONV = (float*) (ws + take((size_t)1152 * 128 * 4));
  __bf16* BCONV = (__bf16*)(ws + take((size_t)1152 * 128 * 2));
  float*  OC    = (float*) (ws + take((size_t)BT_ * 128 * HW_ * 4));
  float*  X1R   = (float*) (ws + take((size_t)BT_ * 64 * HW_ * 4));
  float*  X1I   = (float*) (ws + take((size_t)BT_ * 64 * HW_ * 4));
  __bf16* XTBF  = (__bf16*)(ws + take((size_t)NTOK_ * 128 * 2));
  float*  VINV  = (float*) (ws + take(8192 * 4));
  float*  WPROJ = (float*) (ws + take((size_t)128 * 128 * 4));
  float*  WBACK = (float*) (ws + take((size_t)128 * 128 * 4));
  __bf16* BPROJ = (__bf16*)(ws + take((size_t)128 * 128 * 2));
  __bf16* BBACK = (__bf16*)(ws + take((size_t)128 * 128 * 2));
  float*  XE    = (float*) (ws + take((size_t)NTOK_ * 128 * 4));   // proj out, reused as backproj out
  __bf16* HEBF  = (__bf16*)(ws + take((size_t)NTOK_ * 128 * 2));
  float*  FLATF = (float*) (ws + take((size_t)NTOK_ * 128 * 4));
  __bf16* MIXBF = (__bf16*)(ws + take((size_t)NTOK_ * 128 * 2));
  __bf16* B1F   = (__bf16*)(ws + take((size_t)128 * 512 * 2));
  __bf16* B2F   = (__bf16*)(ws + take((size_t)512 * 128 * 2));
  __bf16* HDNBF = (__bf16*)(ws + take((size_t)NTOK_ * 512 * 2));
  float*  POF   = (float*) (ws + take((size_t)NTOK_ * 128 * 4));
  (void)ws_size; (void)in_sizes; (void)n_in; (void)out_size;

  // --- constants / weights ---
  k_build_fmat <<<16, 256, 0, stream>>>(FF, FG);
  k_build_wconv<<<576, 256, 0, stream>>>(wr, wi, WCONV);
  k_packB      <<<576, 256, 0, stream>>>(WCONV, BCONV, 1152, 128);
  k_invert     <<<1, 256, 66560, stream>>>(V_r, V_i, VINV);
  k_build_cw   <<<64, 256, 0, stream>>>(VINV, VINV + 4096, WPROJ);
  k_build_cw   <<<64, 256, 0, stream>>>(V_r, V_i, WBACK);
  k_packB      <<<64, 256, 0, stream>>>(WPROJ, BPROJ, 128, 128);
  k_packB      <<<64, 256, 0, stream>>>(WBACK, BBACK, 128, 128);
  k_packB      <<<256, 256, 0, stream>>>(w1, B1F, 128, 512);
  k_packB      <<<256, 256, 0, stream>>>(w2, B2F, 512, 128);

  // --- spatial stage ---
  k_norm_spatial<<<512, 256, 0, stream>>>(xr, xi, ns_g, ns_b, XSBF, XSR, XSI);
  k_conv        <<<8192, 256, 0, stream>>>(XSBF, BCONV, br, bi, OC);
  k_fft         <<<2048, 256, 98304, stream>>>(XSR, XSI, FF, FG, specr, speci);
  k_combine1    <<<32768, 256, 0, stream>>>(xr, xi, OC, XSR, XSI, gate, X1R, X1I);

  // --- temporal stage ---
  k_norm_temporal<<<512, 256, 0, stream>>>(X1R, X1I, nt_g, nt_b, XTBF);
  k_gemm<<<dim3(NTOK_ / 16, 1), 256, 0, stream>>>(XTBF, BPROJ, nullptr, XE, nullptr,
                                                  NTOK_, 128, 128, 0);
  k_scan<<<2048, 256, 0, stream>>>(XE, dt, omega, gdec, HEBF);
  k_gemm<<<dim3(NTOK_ / 16, 1), 256, 0, stream>>>(HEBF, BBACK, nullptr, XE, nullptr,
                                                  NTOK_, 128, 128, 0);
  k_combine2<<<32768, 256, 0, stream>>>(XE, X1R, X1I);   // X1 now holds x2

  // --- pooling stage ---
  k_norm_pool<<<512, 256, 0, stream>>>(X1R, X1I, np_g, np_b, FLATF);
  k_mix      <<<16384, 256, 0, stream>>>(FLATF, MIXBF);
  k_gemm<<<dim3(NTOK_ / 16, 4), 256, 0, stream>>>(MIXBF, B1F, b1, nullptr, HDNBF,
                                                  NTOK_, 128, 512, 1);
  k_gemm<<<dim3(NTOK_ / 16, 1), 256, 0, stream>>>(HDNBF, B2F, b2, POF, nullptr,
                                                  NTOK_, 512, 128, 0);
  k_final<<<32768, 256, 0, stream>>>(POF, X1R, X1I, out);
}
